// ESM3CrossAttentionFusion_16887811407944
// MI455X (gfx1250) — matrix operand order repaired and checked by reading, not compiled, not run on hardware
//
#include <hip/hip_runtime.h>

// ---------------------------------------------------------------------------
// ESM3 cross-attention fusion for MI455X (gfx1250), wave32 + WMMA bf16.
// prep: transpose/cast all weights to bf16
// kv  : K [L,128] and V^T [128,L] via WMMA GEMM over ESM_DIM
// attn: flash attention (64 keys/step) + WMMA epilogue (Wo, LayerNorm, gate)
// ---------------------------------------------------------------------------

typedef __attribute__((ext_vector_type(16))) __bf16 v16bf;
typedef __attribute__((ext_vector_type(8)))  __bf16 v8bf;
typedef __attribute__((ext_vector_type(8)))  float  v8f;
typedef __attribute__((ext_vector_type(4)))  int    i32x4;

#define N_ATOMS  16384
#define L_RES    1024
#define ESM_DIM  1280
#define EPS_LN   1e-5f

// workspace byte offsets
#define WS_WQT 0u          // 128x128  bf16 (Wq^T)    32768 B
#define WS_WKT 32768u      // 128x1280 bf16 (Wk^T)   327680 B
#define WS_WVT 360448u     // 128x1280 bf16 (Wv^T)   327680 B
#define WS_WOT 688128u     // 128x128  bf16 (Wo^T)    32768 B
#define WS_WGT 720896u     // 128x256  bf16 (Wg^T)    65536 B
#define WS_KBF 786432u     // 1024x128 bf16 (K)      262144 B
#define WS_VT  1048576u    // 128x1024 bf16 (V^T)    262144 B

#if defined(__has_builtin)
#if __has_builtin(__builtin_amdgcn_global_load_async_to_lds_b128) && \
    __has_builtin(__builtin_amdgcn_s_wait_asynccnt)
#define USE_ASYNC_LDS 1
#endif
#endif

static __device__ __forceinline__ v8f wmma_bf16(v16bf a, v16bf b, v8f c) {
  return __builtin_amdgcn_wmma_f32_16x16x32_bf16(false, a, false, b,
                                                 (short)0, c, false, false);
}

// Build a bf16 A-operand slice from an f32 row pointer (K window of 32).
static __device__ __forceinline__ v16bf a_from_f32(const float* row, int half) {
  v16bf a;
  const float* p0 = row + 16 * half;
  const float* p1 = row + 16 * half + 8;
#pragma unroll
  for (int j = 0; j < 8; ++j) a[j]     = (__bf16)p0[j];
#pragma unroll
  for (int j = 0; j < 8; ++j) a[8 + j] = (__bf16)p1[j];
  return a;
}

// ---------------------------------------------------------------------------
// Kernel 1: cast + transpose all weights to bf16 (tiny).
// ---------------------------------------------------------------------------
__global__ void prep_weights_kernel(const float* __restrict__ Wq,
                                    const float* __restrict__ Wk,
                                    const float* __restrict__ Wv,
                                    const float* __restrict__ Wo,
                                    const float* __restrict__ Wg,
                                    __bf16* __restrict__ WqT,
                                    __bf16* __restrict__ WkT,
                                    __bf16* __restrict__ WvT,
                                    __bf16* __restrict__ WoT,
                                    __bf16* __restrict__ WgT) {
  int i = blockIdx.x * blockDim.x + threadIdx.x;
  if (i < 16384) {                       // WqT[c*128+e] = Wq[e,c]
    int c = i >> 7, e = i & 127;
    WqT[i] = (__bf16)Wq[e * 128 + c];
  } else if (i < 180224) {               // WkT[c*1280+e] = Wk[e,c]
    int j = i - 16384;
    int c = j / ESM_DIM, e = j - c * ESM_DIM;
    WkT[j] = (__bf16)Wk[e * 128 + c];
  } else if (i < 344064) {               // WvT[c*1280+e] = Wv[e,c]
    int j = i - 180224;
    int c = j / ESM_DIM, e = j - c * ESM_DIM;
    WvT[j] = (__bf16)Wv[e * 128 + c];
  } else if (i < 360448) {               // WoT[c*128+e] = Wo[e,c]
    int j = i - 344064;
    int c = j >> 7, e = j & 127;
    WoT[j] = (__bf16)Wo[e * 128 + c];
  } else if (i < 393216) {               // WgT[c*256+e] = Wg[e,c]  (e in 0..255)
    int j = i - 360448;
    int c = j >> 8, e = j & 255;
    WgT[j] = (__bf16)Wg[e * 128 + c];
  }
}

// ---------------------------------------------------------------------------
// Kernel 2: K = esm @ Wk -> Kbf [L,128];  V = esm @ Wv -> Vt [128,L] (transposed)
// One wave per 16x16 tile; 40 bf16 WMMAs over ESM_DIM.
// ---------------------------------------------------------------------------
__global__ void __launch_bounds__(256)
kv_gemm_kernel(const float* __restrict__ esm,
               const __bf16* __restrict__ WkT,
               const __bf16* __restrict__ WvT,
               __bf16* __restrict__ Kbf,
               __bf16* __restrict__ Vt) {
  const int lane = threadIdx.x & 31;
  const int wave = threadIdx.x >> 5;
  const int l16  = lane & 15;
  const int half = lane >> 4;

  const int tile = blockIdx.x * 8 + wave;   // 0..1023
  const int mat  = tile >> 9;               // 0 = K, 1 = V
  const int rem  = tile & 511;
  const int tr   = rem >> 3;                // residue tile (16 rows)
  const int tc   = rem & 7;                 // out-dim tile (16 cols)

  const __bf16* WT   = mat ? WvT : WkT;
  const float*  arow = esm + (tr * 16 + l16) * ESM_DIM;
  const __bf16* brow = WT + (tc * 16 + l16) * ESM_DIM;

  v8f c;
#pragma unroll
  for (int i = 0; i < 8; ++i) c[i] = 0.f;

  for (int kk = 0; kk < ESM_DIM; kk += 32) {
    v16bf a = a_from_f32(arow + kk, half);
    v16bf b = *(const v16bf*)(brow + kk + 16 * half);
    c = wmma_bf16(a, b, c);
  }

  if (mat == 0) {
#pragma unroll
    for (int r = 0; r < 8; ++r)
      Kbf[(tr * 16 + r + 8 * half) * 128 + tc * 16 + l16] = (__bf16)c[r];
  } else {
#pragma unroll
    for (int r = 0; r < 8; ++r)
      Vt[(tc * 16 + l16) * L_RES + tr * 16 + r + 8 * half] = (__bf16)c[r];
  }
}

// ---------------------------------------------------------------------------
// Kernel 3: flash cross-attention + WMMA epilogue.
// Block = 16 atoms, 8 waves = 8 heads (then 8 column-tiles for the epilogue).
// ---------------------------------------------------------------------------
__global__ void __launch_bounds__(256)
attn_kernel(const float* __restrict__ atom_feat,
            const __bf16* __restrict__ WqT,
            const __bf16* __restrict__ Kbf,
            const __bf16* __restrict__ Vt,
            const __bf16* __restrict__ WoT,
            const __bf16* __restrict__ WgT,
            const float* __restrict__ bo,
            const float* __restrict__ gamma, const float* __restrict__ beta,
            const float* __restrict__ bg,
            float* __restrict__ out) {
  __shared__ __align__(32) unsigned char smem[53248];

  const int t    = threadIdx.x;
  const int lane = t & 31;
  const int wave = t >> 5;          // head index / epilogue column tile
  const int l16  = lane & 15;
  const int half = lane >> 4;
  const int n0   = blockIdx.x * 16; // atom tile base
  const int h    = wave;

  __bf16* qtile = (__bf16*)(smem + wave * 512);            // per-wave [16][16]
  __bf16* ptile = (__bf16*)(smem + 4096 + wave * 2048);    // per-wave [16][64]
  float*  ctxs  = (float*)(smem + 20480);                  // [16][128]
  float*  ys    = (float*)(smem + 28672);                  // [16][128]
  float*  onrm  = (float*)(smem + 36864);                  // [16][128]
  float*  afs   = (float*)(smem + 45056);                  // [16][128]

  // stage atom_feat tile for the epilogue (consumed after barrier)
#ifdef USE_ASYNC_LDS
  {
    // 2048 floats = 512 x b128; 256 threads x 2 async copies, overlapped with
    // the whole flash loop; ASYNCcnt drained before the epilogue barrier.
    int e0 = t * 4;
    int e1 = (t + 256) * 4;
    __builtin_amdgcn_global_load_async_to_lds_b128(
        (__attribute__((address_space(1))) i32x4*)(atom_feat + n0 * 128 + e0),
        (__attribute__((address_space(3))) i32x4*)(afs + e0), 0, 0);
    __builtin_amdgcn_global_load_async_to_lds_b128(
        (__attribute__((address_space(1))) i32x4*)(atom_feat + n0 * 128 + e1),
        (__attribute__((address_space(3))) i32x4*)(afs + e1), 0, 0);
  }
#else
#pragma unroll
  for (int j = 0; j < 8; ++j) {
    int e = t + 256 * j;
    afs[e] = atom_feat[n0 * 128 + e];
  }
#endif

  v8f czero;
#pragma unroll
  for (int i = 0; i < 8; ++i) czero[i] = 0.f;

  // ---- Q tile = atom_feat[16,128] @ Wq[:, 16h:16h+16], softmax scale folded
  v8f cq = czero;
  {
    const float*  arow = atom_feat + (n0 + l16) * 128;
    const __bf16* brow = WqT + (h * 16 + l16) * 128;
#pragma unroll
    for (int kk = 0; kk < 128; kk += 32) {
      v16bf a = a_from_f32(arow + kk, half);
      v16bf b = *(const v16bf*)(brow + kk + 16 * half);
      cq = wmma_bf16(a, b, cq);
    }
  }
  // C-layout -> per-wave LDS (with 1/sqrt(D) scale) -> reload in A-layout
#pragma unroll
  for (int r = 0; r < 8; ++r)
    qtile[(r + 8 * half) * 16 + l16] = (__bf16)(0.25f * cq[r]);
  asm volatile("s_wait_dscnt 0" ::: "memory");  // per-wave LDS RAW fence

  v16bf aQ;
  {
    v16bf q16 = *(const v16bf*)(qtile + l16 * 16);
#pragma unroll
    for (int j = 0; j < 16; ++j) aQ[j] = half ? (__bf16)0.f : q16[j];
  }

  // ---- flash attention over L=1024 keys, 64 keys per step
  float mrow[8], srow[8];
#pragma unroll
  for (int r = 0; r < 8; ++r) { mrow[r] = -1e30f; srow[r] = 0.f; }
  v8f acc = czero;

  for (int lb = 0; lb < L_RES; lb += 64) {
    // scores: 4 WMMAs of 16 keys each (upper lanes of B = zero pad for K>=16)
    v8f s[4];
#pragma unroll
    for (int q = 0; q < 4; ++q) {
      v16bf bk = *(const v16bf*)(Kbf + (lb + 16 * q + l16) * 128 + h * 16);
      if (half) {
#pragma unroll
        for (int j = 0; j < 16; ++j) bk[j] = (__bf16)0.f;
      }
      s[q] = wmma_bf16(aQ, bk, czero);
    }

    if (lb + 64 < L_RES) {  // prefetch next K/V blocks (global_prefetch_b8)
      __builtin_prefetch(Kbf + (lb + 64 + l16) * 128 + h * 16, 0, 1);
      __builtin_prefetch(Vt + (h * 16 + l16) * L_RES + lb + 64, 0, 1);
    }

    // online softmax: per C-VGPR r, lanes 0-15 carry row r, 16-31 row r+8
#pragma unroll
    for (int r = 0; r < 8; ++r) {
      float bm = fmaxf(fmaxf(s[0][r], s[1][r]), fmaxf(s[2][r], s[3][r]));
#pragma unroll
      for (int m = 1; m < 16; m <<= 1) bm = fmaxf(bm, __shfl_xor(bm, m, 32));
      float mnew  = fmaxf(mrow[r], bm);
      float alpha = __expf(mrow[r] - mnew);
      mrow[r] = mnew;
      float p0 = __expf(s[0][r] - mnew);
      float p1 = __expf(s[1][r] - mnew);
      float p2 = __expf(s[2][r] - mnew);
      float p3 = __expf(s[3][r] - mnew);
      float rs = (p0 + p1) + (p2 + p3);
#pragma unroll
      for (int m = 1; m < 16; m <<= 1) rs += __shfl_xor(rs, m, 32);
      srow[r] = srow[r] * alpha + rs;
      acc[r] *= alpha;
      __bf16* prow = ptile + (r + 8 * half) * 64 + l16;
      prow[0]  = (__bf16)p0;
      prow[16] = (__bf16)p1;
      prow[32] = (__bf16)p2;
      prow[48] = (__bf16)p3;
    }
    asm volatile("s_wait_dscnt 0" ::: "memory");  // per-wave LDS RAW fence

    // ctx accumulate: P[16,64] x V[64,16] as two K=32 WMMAs
#pragma unroll
    for (int q = 0; q < 2; ++q) {
      v16bf aP;
      {
        v8bf plo = *(const v8bf*)(ptile + l16 * 64 + 32 * q + 16 * half);
        v8bf phi = *(const v8bf*)(ptile + l16 * 64 + 32 * q + 16 * half + 8);
#pragma unroll
        for (int j = 0; j < 8; ++j) { aP[j] = plo[j]; aP[8 + j] = phi[j]; }
      }
      v16bf bv = *(const v16bf*)(Vt + (h * 16 + l16) * L_RES + lb + 32 * q + 16 * half);
      acc = wmma_bf16(aP, bv, acc);
    }
  }

  // normalize and publish ctx head-slice to shared LDS
#pragma unroll
  for (int r = 0; r < 8; ++r) {
    float inv = 1.f / srow[r];
    ctxs[(r + 8 * half) * 128 + h * 16 + l16] = acc[r] * inv;
  }
#ifdef USE_ASYNC_LDS
  __builtin_amdgcn_s_wait_asynccnt(0);  // afs resident before the barrier
#endif
  __syncthreads();

  // ---- epilogue on WMMA: wave w owns output columns c0 = 16w
  const int c0 = wave * 16;

  // y = ctx @ Wo + bo
  v8f cy = czero;
#pragma unroll
  for (int kk = 0; kk < 128; kk += 32) {
    v16bf a = a_from_f32(ctxs + l16 * 128 + kk, half);
    v16bf b = *(const v16bf*)(WoT + (c0 + l16) * 128 + kk + 16 * half);
    cy = wmma_bf16(a, b, cy);
  }
  {
    float bov = bo[c0 + l16];
#pragma unroll
    for (int r = 0; r < 8; ++r)
      ys[(r + 8 * half) * 128 + c0 + l16] = cy[r] + bov;
  }
  __syncthreads();

  // LayerNorm over 128 (one thread per row; trivial cost)
  if (t < 16) {
    float mu = 0.f;
    for (int k = 0; k < 128; ++k) mu += ys[t * 128 + k];
    mu *= (1.f / 128.f);
    float var = 0.f;
    for (int k = 0; k < 128; ++k) {
      float d = ys[t * 128 + k] - mu;
      var += d * d;
    }
    var *= (1.f / 128.f);
    float rs = rsqrtf(var + EPS_LN);
    for (int k = 0; k < 128; ++k)
      onrm[t * 128 + k] = gamma[k] * (ys[t * 128 + k] - mu) * rs + beta[k];
  }
  __syncthreads();

  // gate = sigmoid([atom_feat | out] @ Wg + bg); blend, store
  v8f cg = czero;
#pragma unroll
  for (int kk = 0; kk < 256; kk += 32) {
    const float* src = (kk < 128) ? (afs + l16 * 128 + kk)
                                  : (onrm + l16 * 128 + (kk - 128));
    v16bf a = a_from_f32(src, half);
    v16bf b = *(const v16bf*)(WgT + (c0 + l16) * 256 + kk + 16 * half);
    cg = wmma_bf16(a, b, cg);
  }
  {
    float bgv = bg[c0 + l16];
#pragma unroll
    for (int r = 0; r < 8; ++r) {
      int m = r + 8 * half;
      float sg = 1.f / (1.f + __expf(-(cg[r] + bgv)));
      float on = onrm[m * 128 + c0 + l16];
      float af = afs[m * 128 + c0 + l16];
      out[(n0 + m) * 128 + c0 + l16] = sg * on + (1.f - sg) * af;
    }
  }
}

// ---------------------------------------------------------------------------
extern "C" void kernel_launch(void* const* d_in, const int* in_sizes, int n_in,
                              void* d_out, int out_size, void* d_ws, size_t ws_size,
                              hipStream_t stream) {
  const float* atom_feat = (const float*)d_in[0];
  const float* esm       = (const float*)d_in[1];
  // d_in[2] = atom_res_idx : unused by the reference math
  const float* Wq    = (const float*)d_in[3];
  const float* Wk    = (const float*)d_in[4];
  const float* Wv    = (const float*)d_in[5];
  const float* Wo    = (const float*)d_in[6];
  const float* bo    = (const float*)d_in[7];
  const float* gamma = (const float*)d_in[8];
  const float* beta  = (const float*)d_in[9];
  const float* Wg    = (const float*)d_in[10];
  const float* bg    = (const float*)d_in[11];
  float* out = (float*)d_out;

  unsigned char* ws = (unsigned char*)d_ws;
  __bf16* WqT = (__bf16*)(ws + WS_WQT);
  __bf16* WkT = (__bf16*)(ws + WS_WKT);
  __bf16* WvT = (__bf16*)(ws + WS_WVT);
  __bf16* WoT = (__bf16*)(ws + WS_WOT);
  __bf16* WgT = (__bf16*)(ws + WS_WGT);
  __bf16* Kbf = (__bf16*)(ws + WS_KBF);
  __bf16* Vt  = (__bf16*)(ws + WS_VT);

  prep_weights_kernel<<<1536, 256, 0, stream>>>(Wq, Wk, Wv, Wo, Wg,
                                                WqT, WkT, WvT, WoT, WgT);
  kv_gemm_kernel<<<128, 256, 0, stream>>>(esm, WkT, WvT, Kbf, Vt);
  attn_kernel<<<N_ATOMS / 16, 256, 0, stream>>>(atom_feat, WqT, Kbf, Vt,
                                                WoT, WgT, bo, gamma, beta, bg,
                                                out);
}
